// MultiHeadMLP_31619549233292
// MI455X (gfx1250) — compile-verified, matrix operand order checked by reading, not run on hardware
//
#include <hip/hip_runtime.h>
#include <hip/hip_bf16.h>

// Problem constants (from reference)
#define NN 50000
#define EE 800000
#define DD 128
#define EDD 64
#define HH 8
#define HDD 16
#define EHD 8
#define ATT_S 40           // 2*HD + EHD
#define NEG_SLOPE 0.01f

typedef __attribute__((ext_vector_type(16))) _Float16 v16h;
typedef __attribute__((ext_vector_type(8)))  _Float16 v8h;
typedef __attribute__((ext_vector_type(8)))  float    v8f;

// ---------------- elementwise helpers ----------------

__global__ void k_f2h(const float* __restrict__ in, _Float16* __restrict__ out, int n) {
  int i = blockIdx.x * blockDim.x + threadIdx.x;
  if (i < n) out[i] = (_Float16)in[i];
}

__global__ void k_zero(float* __restrict__ p, int n) {
  int i = blockIdx.x * blockDim.x + threadIdx.x;
  if (i < n) p[i] = 0.0f;
}

__global__ void k_copy_i32(const int* __restrict__ in, int* __restrict__ out, int n) {
  int i = blockIdx.x * blockDim.x + threadIdx.x;
  if (i < n) out[i] = in[i];
}

__global__ void k_aggbias_h(const float* __restrict__ agg, const float* __restrict__ bias,
                            _Float16* __restrict__ out, int n) {
  int i = blockIdx.x * blockDim.x + threadIdx.x;
  if (i < n) out[i] = (_Float16)(agg[i] + bias[i & (DD - 1)]);
}

// ---------------- WMMA GEMM: Y[M x 128] = Ah[M x 128] @ Wh[128 x 128]^T (+bias) ----------------
// 8 waves per block; wave w owns output cols [16w, 16w+16); 16 rows per block.
// A fragment (16x32 f16): lane l -> row l&15; halves 0..7 = K kb..kb+7, halves 8..15 = K kb+16..kb+23,
// kb = kt + (l<16 ? 0 : 8).  B[k][n] = W[n][k] -> same contiguous pattern from row n of W.
__global__ __launch_bounds__(256) void k_gemm128(
    const _Float16* __restrict__ A,   // [M x 128] f16 row-major
    const _Float16* __restrict__ W,   // [128 x 128] f16 row-major (output-major)
    float* __restrict__ Y,            // [M x 128] f32
    const float* __restrict__ bias)   // [128] or nullptr
{
  const int K = DD;
  const int lane = threadIdx.x & 31;
  const int wave = threadIdx.x >> 5;
  const int hl   = lane & 15;
  const int m0   = blockIdx.x * 16;
  const int ncol = wave * 16 + hl;

  v8f c = {};
#pragma unroll
  for (int kt = 0; kt < K; kt += 32) {
    const int kb = kt + ((lane < 16) ? 0 : 8);
    const _Float16* ap = A + (size_t)(m0 + hl) * K + kb;
    const _Float16* bp = W + (size_t)ncol * K + kb;
    v8h a0 = *(const v8h*)ap;
    v8h a1 = *(const v8h*)(ap + 16);
    v8h b0 = *(const v8h*)bp;
    v8h b1 = *(const v8h*)(bp + 16);
    v16h a, b;
#pragma unroll
    for (int i = 0; i < 8; ++i) { a[i] = a0[i]; a[8 + i] = a1[i]; b[i] = b0[i]; b[8 + i] = b1[i]; }
    c = __builtin_amdgcn_wmma_f32_16x16x32_f16(false, a, false, b, (short)0, c, false, false);
  }

  const float badd = bias ? bias[ncol] : 0.0f;
#pragma unroll
  for (int r = 0; r < 8; ++r) {
    const int row = (lane < 16) ? r : (r + 8);
    Y[(size_t)(m0 + row) * K + ncol] = c[r] + badd;
  }
}

// ---------------- WMMA GEMM for edges: Eout[E x 64] = Ah[E x 64] @ Wh[64 x 64]^T, fused s_e ----------------
// 4 waves per block, 16 edges per block. Tile staged in LDS for the per-head attention dot.
__global__ __launch_bounds__(128) void k_gemm_edge(
    const _Float16* __restrict__ A,   // [E x 64] f16
    const _Float16* __restrict__ W,   // [64 x 64] f16 (output-major)
    const float* __restrict__ att,    // [H x 40]
    float* __restrict__ Eout,         // [E x 64] -> d_out region
    float* __restrict__ s_e)          // [E x H]
{
  const int K = EDD;
  __shared__ float tile[16][EDD];

  const int lane = threadIdx.x & 31;
  const int wave = threadIdx.x >> 5;
  const int hl   = lane & 15;
  const int m0   = blockIdx.x * 16;
  const int ncol = wave * 16 + hl;

  v8f c = {};
#pragma unroll
  for (int kt = 0; kt < K; kt += 32) {
    const int kb = kt + ((lane < 16) ? 0 : 8);
    const _Float16* ap = A + (size_t)(m0 + hl) * K + kb;
    const _Float16* bp = W + (size_t)ncol * K + kb;
    v8h a0 = *(const v8h*)ap;
    v8h a1 = *(const v8h*)(ap + 16);
    v8h b0 = *(const v8h*)bp;
    v8h b1 = *(const v8h*)(bp + 16);
    v16h a, b;
#pragma unroll
    for (int i = 0; i < 8; ++i) { a[i] = a0[i]; a[8 + i] = a1[i]; b[i] = b0[i]; b[8 + i] = b1[i]; }
    c = __builtin_amdgcn_wmma_f32_16x16x32_f16(false, a, false, b, (short)0, c, false, false);
  }

#pragma unroll
  for (int r = 0; r < 8; ++r) {
    const int row = (lane < 16) ? r : (r + 8);
    const float v = c[r];
    Eout[(size_t)(m0 + row) * K + ncol] = v;
    tile[row][ncol] = v;
  }
  __syncthreads();

  // 128 threads -> (16 edges) x (8 heads)
  const int le = threadIdx.x >> 3;
  const int h  = threadIdx.x & 7;
  float s = 0.0f;
#pragma unroll
  for (int j = 0; j < EHD; ++j) s += tile[le][h * EHD + j] * att[h * ATT_S + 2 * HDD + j];
  s_e[(size_t)(m0 + le) * HH + h] = s;
}

// ---------------- per-node attention scores ----------------
__global__ void k_score_nodes(const float* __restrict__ x, const float* __restrict__ att,
                              float* __restrict__ s_row, float* __restrict__ s_col) {
  int t = blockIdx.x * blockDim.x + threadIdx.x;
  if (t >= NN * HH) return;
  const int nd = t >> 3, h = t & 7;
  const float* xp = x + (size_t)nd * DD + h * HDD;
  float sr = 0.0f, sc = 0.0f;
#pragma unroll
  for (int j = 0; j < HDD; ++j) {
    sr += xp[j] * att[h * ATT_S + j];
    sc += xp[j] * att[h * ATT_S + HDD + j];
  }
  s_row[t] = sr;
  s_col[t] = sc;
}

// ---------------- softmax pass A: ex = exp(lrelu(.)), denom segment-sum ----------------
__global__ void k_att_a(const int* __restrict__ ei,
                        const float* __restrict__ s_row, const float* __restrict__ s_col,
                        const float* __restrict__ s_e,
                        float* __restrict__ exv, float* __restrict__ denom) {
  int t = blockIdx.x * blockDim.x + threadIdx.x;
  if (t >= EE * HH) return;
  const int e = t >> 3, h = t & 7;
  const int row = ei[e * 2 + 0];
  const int col = ei[e * 2 + 1];
  float al = s_row[row * HH + h] + s_col[col * HH + h] + s_e[t];
  al = (al > 0.0f) ? al : NEG_SLOPE * al;
  const float ex = __expf(al);
  exv[t] = ex;
  atomicAdd(&denom[row * HH + h], ex);
}

// ---------------- softmax pass B: agg[row] += (ex/denom) * x[col] ----------------
__global__ __launch_bounds__(128) void k_att_b(const int* __restrict__ ei,
                                               const float* __restrict__ exv,
                                               const float* __restrict__ denom,
                                               const float* __restrict__ x,
                                               float* __restrict__ agg) {
  const int e = blockIdx.x;
  const int d = threadIdx.x;          // 0..127
  const int h = d >> 4;               // head
  const int row = ei[e * 2 + 0];
  const int col = ei[e * 2 + 1];
  const float w = exv[e * HH + h] / denom[row * HH + h];
  atomicAdd(&agg[(size_t)row * DD + d], w * x[(size_t)col * DD + d]);
}

// ---------------- launch ----------------
extern "C" void kernel_launch(void* const* d_in, const int* in_sizes, int n_in,
                              void* d_out, int out_size, void* d_ws, size_t ws_size,
                              hipStream_t stream) {
  (void)in_sizes; (void)n_in; (void)out_size; (void)ws_size;

  const float* feats     = (const float*)d_in[0];   // [N,128]
  const float* edge_attr = (const float*)d_in[1];   // [E,64]
  const float* W_fc      = (const float*)d_in[2];   // [128,128]
  const float* W_edge    = (const float*)d_in[3];   // [64,64]
  const float* att       = (const float*)d_in[4];   // [8,40,1]
  const float* bias      = (const float*)d_in[5];   // [128]
  const float* W_out     = (const float*)d_in[6];   // [128,128]
  const float* b_out     = (const float*)d_in[7];   // [128]
  const int*   ei        = (const int*)d_in[8];     // [E,2]

  // Output layout: out (N*D f32) | edge_index (E*2 i32-bits) | e (E*64 f32)
  float* out_ptr = (float*)d_out;
  int*   ei_out  = (int*)d_out + (size_t)NN * DD;
  float* e_out   = (float*)d_out + (size_t)NN * DD + (size_t)EE * 2;

  // Workspace carve (~235 MB)
  char* wsp = (char*)d_ws;
  auto carve = [&](size_t bytes) -> void* {
    void* p = (void*)wsp;
    wsp += (bytes + 255) & ~(size_t)255;
    return p;
  };
  _Float16* feats_h = (_Float16*)carve((size_t)NN * DD * 2);
  _Float16* wfc_h   = (_Float16*)carve((size_t)DD * DD * 2);
  _Float16* wedge_h = (_Float16*)carve((size_t)EDD * EDD * 2);
  _Float16* wout_h  = (_Float16*)carve((size_t)DD * DD * 2);
  _Float16* ea_h    = (_Float16*)carve((size_t)EE * EDD * 2);
  _Float16* aggh    = (_Float16*)carve((size_t)NN * DD * 2);
  float* x     = (float*)carve((size_t)NN * DD * 4);
  float* s_row = (float*)carve((size_t)NN * HH * 4);
  float* s_col = (float*)carve((size_t)NN * HH * 4);
  float* s_e   = (float*)carve((size_t)EE * HH * 4);
  float* exv   = (float*)carve((size_t)EE * HH * 4);
  float* denom = (float*)carve((size_t)NN * HH * 4);
  float* agg   = (float*)carve((size_t)NN * DD * 4);

  const int T = 256;
  auto cdiv = [](long n, int b) { return (int)((n + b - 1) / b); };

  // 1) f32 -> f16 conversions
  k_f2h<<<cdiv((long)NN * DD, T), T, 0, stream>>>(feats, feats_h, NN * DD);
  k_f2h<<<cdiv(DD * DD, T), T, 0, stream>>>(W_fc, wfc_h, DD * DD);
  k_f2h<<<cdiv(EDD * EDD, T), T, 0, stream>>>(W_edge, wedge_h, EDD * EDD);
  k_f2h<<<cdiv(DD * DD, T), T, 0, stream>>>(W_out, wout_h, DD * DD);
  k_f2h<<<cdiv((long)EE * EDD, T), T, 0, stream>>>(edge_attr, ea_h, EE * EDD);

  // 2) zero accumulators
  k_zero<<<cdiv(NN * HH, T), T, 0, stream>>>(denom, NN * HH);
  k_zero<<<cdiv((long)NN * DD, T), T, 0, stream>>>(agg, NN * DD);

  // 3) node transform: x = feats @ W_fc^T  (WMMA f16->f32)
  k_gemm128<<<NN / 16, 256, 0, stream>>>(feats_h, wfc_h, x, nullptr);

  // 4) per-node attention scores
  k_score_nodes<<<cdiv(NN * HH, T), T, 0, stream>>>(x, att, s_row, s_col);

  // 5) edge transform + fused s_e; e written straight into d_out
  k_gemm_edge<<<EE / 16, 128, 0, stream>>>(ea_h, wedge_h, att, e_out, s_e);

  // 6) edge_index passthrough output
  k_copy_i32<<<cdiv(EE * 2, T), T, 0, stream>>>(ei, ei_out, EE * 2);

  // 7) softmax pass A (exp + denom segment-sum via f32 atomics)
  k_att_a<<<cdiv((long)EE * HH, T), T, 0, stream>>>(ei, s_row, s_col, s_e, exv, denom);

  // 8) softmax pass B (weighted message scatter)
  k_att_b<<<EE, 128, 0, stream>>>(ei, exv, denom, x, agg);

  // 9) (agg + bias) -> f16, then out = . @ W_out^T + b_out  (WMMA)
  k_aggbias_h<<<cdiv((long)NN * DD, T), T, 0, stream>>>(agg, bias, aggh, NN * DD);
  k_gemm128<<<NN / 16, 256, 0, stream>>>(aggh, wout_h, out_ptr, b_out);
}
